// Model_34213709480363
// MI455X (gfx1250) — compile-verified, matrix operand order checked by reading, not run on hardware
//
#include <hip/hip_runtime.h>

typedef __attribute__((ext_vector_type(16))) _Float16 v16h;
typedef __attribute__((ext_vector_type(8)))  _Float16 v8h;
typedef __attribute__((ext_vector_type(8)))  float    v8f;

#define BN_EPS 1e-5f
#define NB 8          // batch
#define NPTS 2048     // input points

// ---------------------------------------------------------------------------
// FPS: one block per batch. Points + running min-dist live in LDS.
// ---------------------------------------------------------------------------
__global__ __launch_bounds__(256) void fps_kernel(const float* __restrict__ xyz,
                                                  int N, int npoint,
                                                  int* __restrict__ fidx,
                                                  float* __restrict__ nxyz) {
    __shared__ float sx[2048], sy[2048], sz[2048], sd[2048];
    __shared__ float rv[256];
    __shared__ int   ri[256];
    __shared__ int   sFar;
    int b = blockIdx.x, tid = threadIdx.x;
    for (int i = tid; i < N; i += 256) {
        const float* q = xyz + ((size_t)b * N + i) * 3;
        sx[i] = q[0]; sy[i] = q[1]; sz[i] = q[2]; sd[i] = 1e10f;
    }
    if (tid == 0) sFar = 0;
    __syncthreads();
    for (int it = 0; it < npoint; ++it) {
        int far = sFar;
        if (tid == 0) {
            fidx[b * npoint + it] = far;
            float* o = nxyz + ((size_t)b * npoint + it) * 3;
            o[0] = sx[far]; o[1] = sy[far]; o[2] = sz[far];
        }
        float cx = sx[far], cy = sy[far], cz = sz[far];
        float bm = -1.f; int bix = 0;
        for (int i = tid; i < N; i += 256) {
            float dx = sx[i] - cx, dy = sy[i] - cy, dz = sz[i] - cz;
            float d = dx * dx + dy * dy + dz * dz;
            float nd = fminf(sd[i], d);
            sd[i] = nd;
            if (nd > bm) { bm = nd; bix = i; }
        }
        rv[tid] = bm; ri[tid] = bix;
        __syncthreads();
        for (int off = 128; off; off >>= 1) {
            if (tid < off && rv[tid + off] > rv[tid]) { rv[tid] = rv[tid + off]; ri[tid] = ri[tid + off]; }
            __syncthreads();
        }
        if (tid == 0) sFar = ri[0];
        __syncthreads();
    }
}

// ---------------------------------------------------------------------------
// Ball query: one wave32 per (b,s); ballot compaction = ascending indices.
// ---------------------------------------------------------------------------
__global__ __launch_bounds__(256) void ball_query_kernel(const float* __restrict__ xyz, int N,
                                                         const float* __restrict__ nxyz, int S,
                                                         float r2, int ns,
                                                         int* __restrict__ idx) {
    int wave = threadIdx.x >> 5, lane = threadIdx.x & 31;
    int row = blockIdx.x * 8 + wave;          // row = b*S + s
    int b = row / S, s = row % S;
    const float* c = nxyz + ((size_t)b * S + s) * 3;
    float cx = c[0], cy = c[1], cz = c[2];
    int count = 0, first = -1;
    int* out = idx + (size_t)row * ns;
    for (int base = 0; base < N && count < ns; base += 32) {
        int i = base + lane;
        const float* q = xyz + ((size_t)b * N + i) * 3;
        float dx = q[0] - cx, dy = q[1] - cy, dz = q[2] - cz;
        bool inr = (dx * dx + dy * dy + dz * dz) < r2;
        unsigned mask = (unsigned)__ballot(inr);
        if (inr) {
            int rank = __popc(mask & ((1u << lane) - 1u));
            int pos = count + rank;
            if (pos < ns) out[pos] = i;
        }
        if (first < 0 && mask) first = base + (__ffs(mask) - 1);
        count += __popc(mask);
    }
    if (count > ns) count = ns;
    if (first < 0) first = 0;
    for (int p = count + lane; p < ns; p += 32) out[p] = first;
}

// ---------------------------------------------------------------------------
// Weight prep: f32 [cout][cin] -> f16 [cout][cinPad], zero-padded.
// ---------------------------------------------------------------------------
__global__ __launch_bounds__(256) void wpad_kernel(const float* __restrict__ W,
                                                   _Float16* __restrict__ Wp,
                                                   int cin, int cinPad, int total) {
    int i = blockIdx.x * 256 + threadIdx.x;
    if (i >= total) return;
    int o = i / cinPad, k = i % cinPad;
    Wp[i] = (_Float16)((k < cin) ? W[(size_t)o * cin + k] : 0.f);
}

// ---------------------------------------------------------------------------
// Fragment loads: halves 0..7 of v16h = K contiguous at (kb + lhi),
// halves 8..15 = K contiguous at (kb + 16 + lhi).  Two b128 loads each.
// ---------------------------------------------------------------------------
__device__ __forceinline__ v16h frag_from(const _Float16* base, int kb, int lhi) {
    v8h lo = *(const v8h*)(base + kb + lhi);
    v8h hi = *(const v8h*)(base + kb + 16 + lhi);
    return __builtin_shufflevector(lo, hi, 0, 1, 2, 3, 4, 5, 6, 7,
                                   8, 9, 10, 11, 12, 13, 14, 15);
}

// ---------------------------------------------------------------------------
// Fused SA scale: gather -> 3x (WMMA GEMM + BN + ReLU) in LDS -> max-pool.
// Activations LDS-resident, column-major [col][k].
// ---------------------------------------------------------------------------
struct SAParams {
    const float* xyz; const float* feats; const float* nxyz; const int* idx;
    const _Float16* W[3];
    const float* ga[3]; const float* be[3]; const float* me[3]; const float* va[3];
    int C0pad, C1, C2, C3, cin;
    int Nsrc, CF, S, ns, G;
    int stride0, stride1;     // halves per column row in buf0/buf1 (16B aligned)
    float* fout; int coff, CT;
};

__device__ __forceinline__ void mlp_layer(const _Float16* __restrict__ inb, int strideIn,
                                          _Float16* __restrict__ outb, int strideOut,
                                          int cinPad, int cout,
                                          const _Float16* __restrict__ Wp,
                                          const float* __restrict__ sc,
                                          const float* __restrict__ bi,
                                          int wave, int lane) {
    int tiles = (cout >> 4) * 8;               // (cout/16) row-tiles x 8 col-tiles
    int lhi = (lane & 16) ? 8 : 0;
    for (int t = wave; t < tiles; t += 8) {
        int obase = (t >> 3) * 16, nbase = (t & 7) * 16;
        int n = nbase + (lane & 15);
        const _Float16* Wrow = Wp + (size_t)(obase + (lane & 15)) * cinPad;
        const _Float16* Brow = inb + (size_t)n * strideIn;
        v8f acc = {0.f, 0.f, 0.f, 0.f, 0.f, 0.f, 0.f, 0.f};
        for (int kb = 0; kb < cinPad; kb += 32) {
            v16h a = frag_from(Wrow, kb, lhi);
            v16h bf = frag_from(Brow, kb, lhi);
            acc = __builtin_amdgcn_wmma_f32_16x16x32_f16(false, a, false, bf,
                                                         (short)0, acc, false, false);
        }
        int mo = (lane & 16) ? 8 : 0;
        v8h dv;
#pragma unroll
        for (int r = 0; r < 8; ++r) {
            int o = obase + r + mo;
            float vv = acc[r] * sc[o] + bi[o];
            vv = vv > 0.f ? vv : 0.f;
            dv[r] = (_Float16)vv;
        }
        *(v8h*)(outb + (size_t)n * strideOut + obase + mo) = dv;
    }
}

__global__ __launch_bounds__(256) void sa_scale_kernel(SAParams P) {
    extern __shared__ char smem[];
    _Float16* buf0 = (_Float16*)smem;                       // [128][stride0]
    _Float16* buf1 = buf0 + (size_t)128 * P.stride0;        // [128][stride1]
    float* sc = (float*)(buf1 + (size_t)128 * P.stride1);
    float* bi = sc + 256;
    int tid = threadIdx.x, wave = tid >> 5, lane = tid & 31;
    int perB = P.S / P.G;
    int b = blockIdx.x / perB;
    int sbase = (blockIdx.x % perB) * P.G;

    // gather + recentre; consecutive threads take consecutive channels
    // (features are point-major [b][j][c] -> coalesced reads).
    for (int item = tid; item < 128 * P.C0pad; item += 256) {
        int col = item / P.C0pad, c = item % P.C0pad;
        int g = col / P.ns, n = col % P.ns;
        int s = sbase + g;
        int j = P.idx[((size_t)b * P.S + s) * P.ns + n];
        float val = 0.f;
        if (c < 3)
            val = P.xyz[((size_t)b * P.Nsrc + j) * 3 + c] - P.nxyz[((size_t)b * P.S + s) * 3 + c];
        else if (c < P.cin)
            val = P.feats[((size_t)b * P.Nsrc + j) * P.CF + (c - 3)];
        buf0[(size_t)col * P.stride0 + c] = (_Float16)val;
    }
    __syncthreads();

    for (int o = tid; o < P.C1; o += 256) {
        float s_ = P.ga[0][o] * rsqrtf(P.va[0][o] + BN_EPS);
        sc[o] = s_; bi[o] = P.be[0][o] - P.me[0][o] * s_;
    }
    __syncthreads();
    mlp_layer(buf0, P.stride0, buf1, P.stride1, P.C0pad, P.C1, P.W[0], sc, bi, wave, lane);
    __syncthreads();
    for (int o = tid; o < P.C2; o += 256) {
        float s_ = P.ga[1][o] * rsqrtf(P.va[1][o] + BN_EPS);
        sc[o] = s_; bi[o] = P.be[1][o] - P.me[1][o] * s_;
    }
    __syncthreads();
    mlp_layer(buf1, P.stride1, buf0, P.stride0, P.C1, P.C2, P.W[1], sc, bi, wave, lane);
    __syncthreads();
    for (int o = tid; o < P.C3; o += 256) {
        float s_ = P.ga[2][o] * rsqrtf(P.va[2][o] + BN_EPS);
        sc[o] = s_; bi[o] = P.be[2][o] - P.me[2][o] * s_;
    }
    __syncthreads();
    mlp_layer(buf0, P.stride0, buf1, P.stride1, P.C2, P.C3, P.W[2], sc, bi, wave, lane);
    __syncthreads();

    // max-pool over ns samples per centroid; fout point-major [b][s][CT]
    for (int item = tid; item < P.C3 * P.G; item += 256) {
        int c = item / P.G, g = item % P.G;
        const _Float16* colp = buf1 + (size_t)(g * P.ns) * P.stride1 + c;
        float m = -1e30f;
        for (int n = 0; n < P.ns; ++n) m = fmaxf(m, (float)colp[(size_t)n * P.stride1]);
        P.fout[((size_t)b * P.S + sbase + g) * P.CT + P.coff + c] = m;
    }
}

// f2 (point-major f32 [B*128][640]) -> f16, identity layout
__global__ __launch_bounds__(256) void f2_to_f16_kernel(const float* __restrict__ f2,
                                                        _Float16* __restrict__ out) {
    size_t i = (size_t)blockIdx.x * 256 + threadIdx.x;   // 8*128*640 total
    out[i] = (_Float16)f2[i];
}

// ---------------------------------------------------------------------------
// Head GEMM: in/out point-major [n][C]; one 16x16 WMMA tile per wave.
// ---------------------------------------------------------------------------
__global__ __launch_bounds__(256) void head_gemm_kernel(const _Float16* __restrict__ in,
                                                        const _Float16* __restrict__ Wp,
                                                        const float* __restrict__ ga,
                                                        const float* __restrict__ be,
                                                        const float* __restrict__ me,
                                                        const float* __restrict__ va,
                                                        const float* __restrict__ hb,
                                                        int C, int O, int M,
                                                        _Float16* out16, float* out32,
                                                        int doRelu) {
    int wave = threadIdx.x >> 5, lane = threadIdx.x & 31;
    int t = blockIdx.x * 8 + wave;
    int ntiles = M >> 4;
    int obase = (t / ntiles) * 16, nbase = (t % ntiles) * 16;
    int n = nbase + (lane & 15);
    int lhi = (lane & 16) ? 8 : 0;
    const _Float16* Wrow = Wp + (size_t)(obase + (lane & 15)) * C;
    const _Float16* Brow = in + (size_t)n * C;
    v8f acc = {0.f, 0.f, 0.f, 0.f, 0.f, 0.f, 0.f, 0.f};
    for (int kb = 0; kb < C; kb += 32) {
        v16h a = frag_from(Wrow, kb, lhi);
        v16h bf = frag_from(Brow, kb, lhi);
        acc = __builtin_amdgcn_wmma_f32_16x16x32_f16(false, a, false, bf,
                                                     (short)0, acc, false, false);
    }
    int mo = (lane & 16) ? 8 : 0;
    if (out16) {
        v8h dv;
#pragma unroll
        for (int r = 0; r < 8; ++r) {
            int o = obase + r + mo;
            float s_ = ga[o] * rsqrtf(va[o] + BN_EPS);
            float bb = (hb[o] - me[o]) * s_ + be[o];
            float vv = acc[r] * s_ + bb;
            if (doRelu) vv = vv > 0.f ? vv : 0.f;
            dv[r] = (_Float16)vv;
        }
        *(v8h*)(out16 + (size_t)n * O + obase + mo) = dv;
    } else {
        int b_ = n >> 7, p_ = n & 127;
#pragma unroll
        for (int r = 0; r < 8; ++r) {
            int o = obase + r + mo;
            float s_ = ga[o] * rsqrtf(va[o] + BN_EPS);
            float bb = (hb[o] - me[o]) * s_ + be[o];
            out32[((size_t)b_ * O + o) * 128 + p_] = acc[r] * s_ + bb;
        }
    }
}

// softmax over points axis (128) per (b, channel) row; one wave per row
__global__ __launch_bounds__(256) void softmax_kernel(float* __restrict__ logits) {
    int wave = threadIdx.x >> 5, lane = threadIdx.x & 31;
    int row = blockIdx.x * 8 + wave;                 // 1024 rows
    float* r = logits + (size_t)row * 128;
    float v[4]; float mx = -1e30f;
#pragma unroll
    for (int i = 0; i < 4; ++i) { v[i] = r[lane + 32 * i]; mx = fmaxf(mx, v[i]); }
    for (int m = 16; m; m >>= 1) mx = fmaxf(mx, __shfl_xor(mx, m, 32));
    float sum = 0.f;
#pragma unroll
    for (int i = 0; i < 4; ++i) { v[i] = __expf(v[i] - mx); sum += v[i]; }
    for (int m = 16; m; m >>= 1) sum += __shfl_xor(sum, m, 32);
    float inv = 1.f / sum;
#pragma unroll
    for (int i = 0; i < 4; ++i) r[lane + 32 * i] = v[i] * inv;
}

// pred[b,s,:] = sum_p h[b,s,p] * xyz2[b,p,:]
__global__ __launch_bounds__(128) void pred_kernel(const float* __restrict__ h,
                                                   const float* __restrict__ xyz2,
                                                   float* __restrict__ pred) {
    int b = blockIdx.x, s = threadIdx.x;
    const float* hr = h + ((size_t)b * 128 + s) * 128;
    float ax = 0.f, ay = 0.f, az = 0.f;
    for (int p = 0; p < 128; ++p) {
        float w = hr[p];
        const float* q = xyz2 + ((size_t)b * 128 + p) * 3;
        ax += w * q[0]; ay += w * q[1]; az += w * q[2];
    }
    float* o = pred + ((size_t)b * 128 + s) * 3;
    o[0] = ax; o[1] = ay; o[2] = az;
}

__global__ void zero_scalar_kernel(float* p) { *p = 0.f; }

// symmetric Chamfer; gt + pred staged in LDS
__global__ __launch_bounds__(256) void chamfer_kernel(const float* __restrict__ pred,
                                                      const float* __restrict__ gt,
                                                      float* __restrict__ out) {
    __shared__ float gx[2048], gy[2048], gz[2048];
    __shared__ float px[128], py[128], pz[128];
    __shared__ float d1s[128];
    __shared__ float red[256];
    int b = blockIdx.x, tid = threadIdx.x;
    for (int i = tid; i < 2048; i += 256) {
        const float* q = gt + ((size_t)b * 2048 + i) * 3;
        gx[i] = q[0]; gy[i] = q[1]; gz[i] = q[2];
    }
    if (tid < 128) {
        const float* q = pred + ((size_t)b * 128 + tid) * 3;
        px[tid] = q[0]; py[tid] = q[1]; pz[tid] = q[2];
    }
    __syncthreads();
    float d2sum = 0.f;
    for (int i = tid; i < 2048; i += 256) {
        float qx = gx[i], qy = gy[i], qz = gz[i], mn = 1e30f;
        for (int s = 0; s < 128; ++s) {
            float dx = px[s] - qx, dy = py[s] - qy, dz = pz[s] - qz;
            mn = fminf(mn, dx * dx + dy * dy + dz * dz);
        }
        d2sum += mn;
    }
    if (tid < 128) {
        float qx = px[tid], qy = py[tid], qz = pz[tid], mn = 1e30f;
        for (int i = 0; i < 2048; ++i) {
            float dx = gx[i] - qx, dy = gy[i] - qy, dz = gz[i] - qz;
            mn = fminf(mn, dx * dx + dy * dy + dz * dz);
        }
        d1s[tid] = mn;
    }
    red[tid] = d2sum;
    __syncthreads();
    for (int off = 128; off; off >>= 1) {
        if (tid < off) red[tid] += red[tid + off];
        __syncthreads();
    }
    if (tid == 0) {
        float s1 = 0.f;
        for (int s = 0; s < 128; ++s) s1 += d1s[s];
        float cd = s1 / 128.f + red[0] / 2048.f;
        atomicAdd(out, cd / (float)NB);
    }
}

// ---------------------------------------------------------------------------
// Host
// ---------------------------------------------------------------------------
struct HLayer { const float *W, *ga, *be, *me, *va, *bi; };

extern "C" void kernel_launch(void* const* d_in, const int* in_sizes, int n_in,
                              void* d_out, int out_size, void* d_ws, size_t ws_size,
                              hipStream_t stream) {
    (void)in_sizes; (void)n_in; (void)out_size; (void)ws_size;
    int p = 0;
    const float* pc = (const float*)d_in[p++];
    const float* gt = (const float*)d_in[p++];
    HLayer sa1L[3][3], sa2L[3][3], headL[4];
    auto grab = [&](bool bias) {
        HLayer L;
        L.W  = (const float*)d_in[p++]; L.ga = (const float*)d_in[p++];
        L.be = (const float*)d_in[p++]; L.me = (const float*)d_in[p++];
        L.va = (const float*)d_in[p++];
        L.bi = bias ? (const float*)d_in[p++] : nullptr;
        return L;
    };
    for (int s = 0; s < 3; ++s) for (int l = 0; l < 3; ++l) sa1L[s][l] = grab(false);
    for (int s = 0; s < 3; ++s) for (int l = 0; l < 3; ++l) sa2L[s][l] = grab(false);
    for (int l = 0; l < 4; ++l) headL[l] = grab(true);

    // channel specs
    const int sa1C[3][3] = {{32, 32, 64}, {64, 64, 128}, {64, 96, 128}};
    const int sa2C[3][3] = {{64, 64, 128}, {128, 128, 256}, {128, 128, 256}};
    const int headCO[4][2] = {{640, 512}, {512, 256}, {256, 128}, {128, 128}};

    // workspace carve-out
    size_t off = 0;
    auto alloc = [&](size_t bytes) {
        void* q = (char*)d_ws + off;
        off = (off + bytes + 255) & ~(size_t)255;
        return q;
    };
    int*      fidx1  = (int*)alloc(NB * 512 * sizeof(int));
    float*    nxyz1  = (float*)alloc(NB * 512 * 3 * sizeof(float));
    int*      fidx2  = (int*)alloc(NB * 128 * sizeof(int));
    float*    nxyz2  = (float*)alloc(NB * 128 * 3 * sizeof(float));
    int*      idxbuf = (int*)alloc((size_t)NB * 512 * 128 * sizeof(int));
    float*    f1     = (float*)alloc((size_t)NB * 512 * 320 * sizeof(float)); // [b][s][c]
    float*    f2     = (float*)alloc((size_t)NB * 128 * 640 * sizeof(float)); // [b][s][c]
    _Float16* h16a   = (_Float16*)alloc((size_t)1024 * 640 * sizeof(_Float16)); // [n][C]
    _Float16* h16b   = (_Float16*)alloc((size_t)1024 * 512 * sizeof(_Float16));
    float*    logits = (float*)alloc((size_t)NB * 128 * 128 * sizeof(float));

    // padded f16 weights
    auto prep = [&](const float* W, int cout, int cin, int cinPad) {
        _Float16* Wp = (_Float16*)alloc((size_t)cout * cinPad * sizeof(_Float16));
        int total = cout * cinPad;
        wpad_kernel<<<(total + 255) / 256, 256, 0, stream>>>(W, Wp, cin, cinPad, total);
        return (const _Float16*)Wp;
    };
    const _Float16 *sa1Wp[3][3], *sa2Wp[3][3], *headWp[4];
    for (int s = 0; s < 3; ++s) {
        int cin = 3;
        for (int l = 0; l < 3; ++l) {
            int cinPad = (cin + 31) & ~31;
            sa1Wp[s][l] = prep(sa1L[s][l].W, sa1C[s][l], cin, cinPad);
            cin = sa1C[s][l];
        }
    }
    for (int s = 0; s < 3; ++s) {
        int cin = 323;
        for (int l = 0; l < 3; ++l) {
            int cinPad = (cin + 31) & ~31;
            sa2Wp[s][l] = prep(sa2L[s][l].W, sa2C[s][l], cin, cinPad);
            cin = sa2C[s][l];
        }
    }
    for (int l = 0; l < 4; ++l)
        headWp[l] = prep(headL[l].W, headCO[l][1], headCO[l][0], headCO[l][0]);

    auto launch_sa = [&](const float* xyz, int Nsrc, const float* feats, int CF,
                         const float* nxyz, int S, float radius, int ns,
                         HLayer* Ls, const _Float16* const* Wp,
                         int C1, int C2, int C3,
                         float* fout, int coff, int CT) {
        ball_query_kernel<<<NB * S / 8, 256, 0, stream>>>(xyz, Nsrc, nxyz, S,
                                                          radius * radius, ns, idxbuf);
        SAParams P;
        P.xyz = xyz; P.feats = feats; P.nxyz = nxyz; P.idx = idxbuf;
        for (int l = 0; l < 3; ++l) {
            P.W[l] = Wp[l];
            P.ga[l] = Ls[l].ga; P.be[l] = Ls[l].be; P.me[l] = Ls[l].me; P.va[l] = Ls[l].va;
        }
        P.cin = 3 + CF;
        P.C0pad = (P.cin + 31) & ~31;
        P.C1 = C1; P.C2 = C2; P.C3 = C3;
        P.Nsrc = Nsrc; P.CF = CF; P.S = S; P.ns = ns; P.G = 128 / ns;
        int rows0 = P.C0pad > C2 ? P.C0pad : C2;
        int rows1 = C1 > C3 ? C1 : C3;
        P.stride0 = rows0 + 8;                 // 16B-aligned, bank-spread
        P.stride1 = rows1 + 8;
        P.fout = fout; P.coff = coff; P.CT = CT;
        size_t lds = (size_t)128 * (P.stride0 + P.stride1) * sizeof(_Float16)
                   + 2 * 256 * sizeof(float);
        sa_scale_kernel<<<NB * S / P.G, 256, lds, stream>>>(P);
    };

    // ---- SA1: 2048 pts -> 512 centroids, out 320 channels ----
    fps_kernel<<<NB, 256, 0, stream>>>(pc, NPTS, 512, fidx1, nxyz1);
    {
        const float r[3] = {0.1f, 0.2f, 0.4f};
        const int ns[3] = {16, 32, 128};
        const int coff[3] = {0, 64, 192};
        for (int k = 0; k < 3; ++k)
            launch_sa(pc, NPTS, nullptr, 0, nxyz1, 512, r[k], ns[k],
                      sa1L[k], sa1Wp[k], sa1C[k][0], sa1C[k][1], sa1C[k][2],
                      f1, coff[k], 320);
    }
    // ---- SA2: 512 pts (+320 feats) -> 128 centroids, out 640 channels ----
    fps_kernel<<<NB, 256, 0, stream>>>(nxyz1, 512, 128, fidx2, nxyz2);
    {
        const float r[3] = {0.2f, 0.4f, 0.8f};
        const int ns[3] = {32, 64, 128};
        const int coff[3] = {0, 128, 384};
        for (int k = 0; k < 3; ++k)
            launch_sa(nxyz1, 512, f1, 320, nxyz2, 128, r[k], ns[k],
                      sa2L[k], sa2Wp[k], sa2C[k][0], sa2C[k][1], sa2C[k][2],
                      f2, coff[k], 640);
    }

    // ---- Head: 640 -> 512 -> 256 -> 128 -> 128(softmax) over M = B*128 ----
    f2_to_f16_kernel<<<(NB * 128 * 640) / 256, 256, 0, stream>>>(f2, h16a);
    const int M = NB * 128;
    head_gemm_kernel<<<(512 / 16) * (M / 16) / 8, 256, 0, stream>>>(
        h16a, headWp[0], headL[0].ga, headL[0].be, headL[0].me, headL[0].va,
        headL[0].bi, 640, 512, M, h16b, nullptr, 1);
    head_gemm_kernel<<<(256 / 16) * (M / 16) / 8, 256, 0, stream>>>(
        h16b, headWp[1], headL[1].ga, headL[1].be, headL[1].me, headL[1].va,
        headL[1].bi, 512, 256, M, h16a, nullptr, 1);
    head_gemm_kernel<<<(128 / 16) * (M / 16) / 8, 256, 0, stream>>>(
        h16a, headWp[2], headL[2].ga, headL[2].be, headL[2].me, headL[2].va,
        headL[2].bi, 256, 128, M, h16b, nullptr, 1);
    head_gemm_kernel<<<(128 / 16) * (M / 16) / 8, 256, 0, stream>>>(
        h16b, headWp[3], headL[3].ga, headL[3].be, headL[3].me, headL[3].va,
        headL[3].bi, 128, 128, M, nullptr, logits, 0);
    softmax_kernel<<<(NB * 128) / 8, 256, 0, stream>>>(logits);

    // ---- pred + chamfer ----
    float* pred = (float*)d_out;
    float* cd   = pred + NB * 128 * 3;
    pred_kernel<<<NB, 128, 0, stream>>>(logits, nxyz2, pred);
    zero_scalar_kernel<<<1, 1, 0, stream>>>(cd);
    chamfer_kernel<<<NB, 256, 0, stream>>>(pred, gt, cd);
}
